// Block_77369540870380
// MI455X (gfx1250) — matrix operand order repaired and checked by reading, not compiled, not run on hardware
//
#include <hip/hip_runtime.h>

typedef __attribute__((ext_vector_type(16))) __bf16          v16bf;
typedef __attribute__((ext_vector_type(16))) unsigned short  v16u;
typedef __attribute__((ext_vector_type(8)))  unsigned short  v8u;
typedef __attribute__((ext_vector_type(8)))  float           v8f;
typedef __attribute__((ext_vector_type(4)))  int             v4i;

#define AS1 __attribute__((address_space(1)))
#define AS3 __attribute__((address_space(3)))

namespace cfg {
constexpr int B = 2, T = 2048, C = 1024, H = 16, D = 64, DFF = 4096;
constexpr int BT = B * T;
}

__device__ __forceinline__ unsigned short f2bf(float f) {
  union { float f; unsigned u; } v; v.f = f;
  unsigned u = v.u;
  unsigned r = u + 0x7FFFu + ((u >> 16) & 1u);   // round-to-nearest-even
  return (unsigned short)(r >> 16);
}

__device__ __forceinline__ v8f wmma_bf16(v16u a, v16u b, v8f c) {
  return __builtin_amdgcn_wmma_f32_16x16x32_bf16(
      false, __builtin_bit_cast(v16bf, a),
      false, __builtin_bit_cast(v16bf, b),
      (short)0, c, false, false);
}

__device__ __forceinline__ v16u load16(const unsigned short* p0,
                                       const unsigned short* p1) {
  v16u r;
  *(v8u*)&r       = *(const v8u*)p0;
  *((v8u*)&r + 1) = *(const v8u*)p1;
  return r;
}

// Async global->LDS 16B copy (per lane), ASYNCcnt-tracked.
__device__ __forceinline__ void async_b128(const unsigned short* g,
                                           unsigned short* l) {
#if __has_builtin(__builtin_amdgcn_global_load_async_to_lds_b128)
  AS1 v4i* gp = (AS1 v4i*)(v4i*)const_cast<unsigned short*>(g);
  AS3 v4i* lp = (AS3 v4i*)(v4i*)l;
  __builtin_amdgcn_global_load_async_to_lds_b128(gp, lp, 0, 0);
#else
  unsigned lofs = (unsigned)(unsigned long long)(void*)l;
  asm volatile("global_load_async_to_lds_b128 %0, %1, off"
               :: "v"(lofs), "v"((unsigned long long)g) : "memory");
#endif
}

__device__ __forceinline__ void wait_asynccnt0() {
#if __has_builtin(__builtin_amdgcn_s_wait_asynccnt)
  __builtin_amdgcn_s_wait_asynccnt(0);
#else
  asm volatile("s_wait_asynccnt 0" ::: "memory");
#endif
}

// ---------------------------------------------------------------------------
// Batched fp32 -> bf16 transpose: out[b][s][r] = in[b][r][s]
// ---------------------------------------------------------------------------
__global__ __launch_bounds__(256) void transpose_cvt(
    const float* __restrict__ in, unsigned short* __restrict__ out,
    int R, int S) {
  __shared__ float tile[32][33];
  const size_t base = (size_t)blockIdx.z * R * S;
  const float* ip = in + base;
  unsigned short* op = out + base;
  int r0 = blockIdx.y * 32, s0 = blockIdx.x * 32;
  int tx = threadIdx.x & 31, ty = threadIdx.x >> 5;  // 32 x 8
#pragma unroll
  for (int k = 0; k < 32; k += 8) {
    int r = r0 + ty + k, s = s0 + tx;
    if (r < R && s < S) tile[ty + k][tx] = ip[(size_t)r * S + s];
  }
  __syncthreads();
#pragma unroll
  for (int k = 0; k < 32; k += 8) {
    int s = s0 + ty + k, r = r0 + tx;
    if (r < R && s < S) op[(size_t)s * R + r] = f2bf(tile[tx][ty + k]);
  }
}

// ---------------------------------------------------------------------------
// LayerNorm (fp32 in) -> bf16 out, one block per row of C=1024
// ---------------------------------------------------------------------------
__global__ __launch_bounds__(256) void ln_kernel(
    const float* __restrict__ x, const float* __restrict__ g,
    const float* __restrict__ bta, unsigned short* __restrict__ out) {
  using namespace cfg;
  __shared__ float red[256];
  const int row = blockIdx.x;
  const float* xr = x + (size_t)row * C;
  float v[4], s = 0.f, s2 = 0.f;
#pragma unroll
  for (int i = 0; i < 4; ++i) {
    v[i] = xr[threadIdx.x + i * 256];
    s += v[i];
    s2 += v[i] * v[i];
  }
  red[threadIdx.x] = s;
  __syncthreads();
  for (int o = 128; o > 0; o >>= 1) {
    if ((int)threadIdx.x < o) red[threadIdx.x] += red[threadIdx.x + o];
    __syncthreads();
  }
  float mu = red[0] * (1.0f / C);
  __syncthreads();
  red[threadIdx.x] = s2;
  __syncthreads();
  for (int o = 128; o > 0; o >>= 1) {
    if ((int)threadIdx.x < o) red[threadIdx.x] += red[threadIdx.x + o];
    __syncthreads();
  }
  float var = red[0] * (1.0f / C) - mu * mu;
  float rstd = rsqrtf(var + 1e-5f);
#pragma unroll
  for (int i = 0; i < 4; ++i) {
    int c = threadIdx.x + i * 256;
    out[(size_t)row * C + c] = f2bf((v[i] - mu) * rstd * g[c] + bta[c]);
  }
}

// ---------------------------------------------------------------------------
// bf16 WMMA GEMM:  acc[m,n] = sum_k A[m,k] * Bt[n,k]   (fp32 acc)
// A: M x K row-major bf16, Bt: N x K row-major bf16 (B transposed).
// Block = 8 waves covering a 128x64 tile: all waves share one 64x32 Bt tile,
// staged in LDS via double-buffered GLOBAL_LOAD_ASYNC_TO_LDS_B128.
// ---------------------------------------------------------------------------
enum { EPI_QK = 0, EPI_VT = 1, EPI_BIAS_RES_F32 = 2, EPI_RELU_BF16 = 3 };

template <int M, int N, int K, int EPI>
__global__ __launch_bounds__(256) void gemm_bf16(
    const unsigned short* __restrict__ A, const unsigned short* __restrict__ Bt,
    const float* __restrict__ bias, const float* __restrict__ res,
    void* __restrict__ outp) {
  using namespace cfg;
  static_assert(M % 128 == 0 && N % 64 == 0 && K % 32 == 0, "tile divisibility");
  constexpr int NT = N / 64;
  constexpr int MB = M / 128;
  constexpr int PS = 40;  // padded LDS row stride (ushorts): 80B, 16B-aligned
  __shared__ __align__(16) unsigned short btile[2][64 * PS];

  const int wid = threadIdx.x >> 5;
  const int lane = threadIdx.x & 31;
  const int half = lane >> 4, l16 = lane & 15;
  const int bm = blockIdx.x % MB;
  const int nt = blockIdx.x / MB;
  const int mt = bm * 8 + wid;

  const unsigned short* ap = A + (size_t)(mt * 16 + l16) * K;

  // staging addresses: wave wid copies Bt rows [wid*8, wid*8+8), 16B per lane
  const int srow = wid * 8 + (lane >> 2);
  const int schunk = (lane & 3) * 8;  // ushort offset within 32-elem K-slice
  const unsigned short* sgp = Bt + (size_t)(nt * 64 + srow) * K + schunk;
  unsigned short* sl0 = &btile[0][srow * PS + schunk];
  unsigned short* sl1 = &btile[1][srow * PS + schunk];

  v8f acc[4] = {};
  async_b128(sgp, sl0);  // prologue: stage k0 = 0 into buffer 0
  int cur = 0;
  for (int k0 = 0; k0 < K; k0 += 32) {
    wait_asynccnt0();   // our slice of the current tile has landed
    __syncthreads();    // everyone's slice has landed
    if (k0 + 32 < K) async_b128(sgp + k0 + 32, cur ? sl0 : sl1);
    __builtin_prefetch(ap + k0 + 64, 0, 0);  // prefetch A two steps ahead
    v16u av = load16(ap + k0 + half * 8, ap + k0 + 16 + half * 8);
    const unsigned short* bb = &btile[cur][0];
#pragma unroll
    for (int g = 0; g < 4; ++g) {
      const unsigned short* pr = bb + (g * 16 + l16) * PS + half * 8;
      acc[g] = wmma_bf16(av, load16(pr, pr + 16), acc[g]);
    }
    __syncthreads();    // reads done before this buffer is refilled
    cur ^= 1;
  }

#pragma unroll
  for (int g = 0; g < 4; ++g) {
#pragma unroll
    for (int j = 0; j < 8; ++j) {
      const int m = mt * 16 + j + half * 8;
      const int n = nt * 64 + g * 16 + l16;
      float val = acc[g][j];
      if constexpr (EPI == EPI_QK) {
        int b = m / T, t = m % T, h = n / D, d = n % D;
        ((unsigned short*)outp)[(((size_t)(b * H + h) * T) + t) * D + d] =
            f2bf(val);
      } else if constexpr (EPI == EPI_VT) {
        int b = m / T, t = m % T, h = n / D, d = n % D;
        ((unsigned short*)outp)[(((size_t)(b * H + h) * D) + d) * T + t] =
            f2bf(val);
      } else if constexpr (EPI == EPI_BIAS_RES_F32) {
        ((float*)outp)[(size_t)m * N + n] =
            val + bias[n] + res[(size_t)m * N + n];
      } else {  // EPI_RELU_BF16
        float r = val + bias[n];
        ((unsigned short*)outp)[(size_t)m * N + n] = f2bf(r > 0.f ? r : 0.f);
      }
    }
  }
}

// ---------------------------------------------------------------------------
// Flash attention: one wave per (b, h, 16-query tile). Causal, online softmax.
// q,k: [B,H,T,D] bf16   vT: [B,H,D,T] bf16   o: [B*T, C] bf16 (head-concat)
// ---------------------------------------------------------------------------
__global__ __launch_bounds__(256) void attn_kernel(
    const unsigned short* __restrict__ q, const unsigned short* __restrict__ k,
    const unsigned short* __restrict__ vt, unsigned short* __restrict__ o) {
  using namespace cfg;
  constexpr int PS = 40;  // padded LDS row stride (ushorts), 16B-aligned rows
  __shared__ __align__(16) unsigned short pbuf[8][16 * PS];
  const int wid = threadIdx.x >> 5;
  const int wave = blockIdx.x * 8 + wid;
  if (wave >= B * H * (T / 16)) return;
  const int lane = threadIdx.x & 31;
  const int half = lane >> 4, l16 = lane & 15;
  const int tt = wave % (T / 16);
  const int bh = wave / (T / 16);
  const unsigned short* qp = q + ((size_t)bh * T + tt * 16 + l16) * D;
  const unsigned short* kp = k + (size_t)bh * T * D;
  const unsigned short* vp = vt + (size_t)bh * D * T;
  unsigned short* pb = &pbuf[wid][0];

  // q fragments for K(=d) steps 0 and 32
  v16u a0 = load16(qp + half * 8, qp + 16 + half * 8);
  v16u a1 = load16(qp + 32 + half * 8, qp + 48 + half * 8);

  float mrow[8], lrow[8];
  v8f acc[4] = {};
#pragma unroll
  for (int j = 0; j < 8; ++j) { mrow[j] = -__builtin_inff(); lrow[j] = 0.f; }

  const int rowbase = tt * 16 + half * 8;
  const int tend = tt * 16 + 15;
  for (int s0 = 0; s0 <= tend; s0 += 32) {
    // scores for 32 keys = two 16-key tiles
    v8f c0 = {}, c1 = {};
    {
      const unsigned short* kr = kp + (size_t)(s0 + l16) * D;
      c0 = wmma_bf16(a0, load16(kr + half * 8, kr + 16 + half * 8), c0);
      c0 = wmma_bf16(a1, load16(kr + 32 + half * 8, kr + 48 + half * 8), c0);
    }
    {
      const unsigned short* kr = kp + (size_t)(s0 + 16 + l16) * D;
      c1 = wmma_bf16(a0, load16(kr + half * 8, kr + 16 + half * 8), c1);
      c1 = wmma_bf16(a1, load16(kr + 32 + half * 8, kr + 48 + half * 8), c1);
    }
    // scale + causal mask + online softmax update (per output row j)
#pragma unroll
    for (int j = 0; j < 8; ++j) {
      const int row = rowbase + j;
      float s0v = c0[j] * 0.03125f;  // 1/sqrt(C), C=1024
      float s1v = c1[j] * 0.03125f;
      if (s0 + l16 > row) s0v = -__builtin_inff();
      if (s0 + 16 + l16 > row) s1v = -__builtin_inff();
      float mx = fmaxf(s0v, s1v);
#pragma unroll
      for (int xm = 1; xm < 16; xm <<= 1)
        mx = fmaxf(mx, __shfl_xor(mx, xm, 32));
      float mnew = fmaxf(mrow[j], mx);
      float alpha = __expf(mrow[j] - mnew);
      float p0 = __expf(s0v - mnew);
      float p1 = __expf(s1v - mnew);
      float rs = p0 + p1;
#pragma unroll
      for (int xm = 1; xm < 16; xm <<= 1) rs += __shfl_xor(rs, xm, 32);
      lrow[j] = lrow[j] * alpha + rs;
      mrow[j] = mnew;
#pragma unroll
      for (int g = 0; g < 4; ++g) acc[g][j] *= alpha;
      const int r = j + half * 8;
      pb[r * PS + l16] = f2bf(p0);
      pb[r * PS + 16 + l16] = f2bf(p1);
    }
    // cross-lane LDS transpose: ensure stores land before fragment reads
    asm volatile("s_wait_dscnt 0" ::: "memory");
    const unsigned short* pr = pb + l16 * PS;
    v16u pv = load16(pr + half * 8, pr + 16 + half * 8);
#pragma unroll
    for (int g = 0; g < 4; ++g) {
      const unsigned short* vr =
          vp + (size_t)(g * 16 + l16) * T + s0 + half * 8;
      acc[g] = wmma_bf16(pv, load16(vr, vr + 16), acc[g]);
    }
    asm volatile("s_wait_dscnt 0" ::: "memory");  // reads done before overwrite
  }

  const int bb = bh / H, hh = bh % H;
#pragma unroll
  for (int j = 0; j < 8; ++j) {
    const float inv = 1.0f / lrow[j];
    const int t = tt * 16 + j + half * 8;
#pragma unroll
    for (int g = 0; g < 4; ++g) {
      const int d = g * 16 + l16;
      o[((size_t)(bb * T + t)) * C + hh * D + d] = f2bf(acc[g][j] * inv);
    }
  }
}

// ---------------------------------------------------------------------------
extern "C" void kernel_launch(void* const* d_in, const int* in_sizes, int n_in,
                              void* d_out, int out_size, void* d_ws,
                              size_t ws_size, hipStream_t stream) {
  using namespace cfg;
  (void)in_sizes; (void)n_in; (void)out_size; (void)ws_size;
  const float* x     = (const float*)d_in[0];
  const float* Wq    = (const float*)d_in[1];
  const float* Wk    = (const float*)d_in[2];
  const float* Wv    = (const float*)d_in[3];
  const float* Wp    = (const float*)d_in[4];
  const float* bp    = (const float*)d_in[5];
  const float* W1    = (const float*)d_in[6];
  const float* b1    = (const float*)d_in[7];
  const float* W2    = (const float*)d_in[8];
  const float* b2    = (const float*)d_in[9];
  const float* ln1_g = (const float*)d_in[10];
  const float* ln1_b = (const float*)d_in[11];
  const float* ln2_g = (const float*)d_in[12];
  const float* ln2_b = (const float*)d_in[13];
  float* out = (float*)d_out;

  size_t off = 0;
  auto carve = [&](size_t bytes) {
    void* p = (char*)d_ws + off;
    off += (bytes + 255) & ~(size_t)255;
    return p;
  };
  unsigned short* WqT = (unsigned short*)carve((size_t)H * D * C * 2);
  unsigned short* WkT = (unsigned short*)carve((size_t)H * D * C * 2);
  unsigned short* WvT = (unsigned short*)carve((size_t)H * D * C * 2);
  unsigned short* WpT = (unsigned short*)carve((size_t)C * C * 2);
  unsigned short* W1T = (unsigned short*)carve((size_t)C * DFF * 2);
  unsigned short* W2T = (unsigned short*)carve((size_t)DFF * C * 2);
  float*          x2  = (float*)carve((size_t)BT * C * 4);
  unsigned short* ob  = (unsigned short*)carve((size_t)BT * C * 2);
  unsigned short* h2b = (unsigned short*)carve((size_t)BT * C * 2);
  // These four are contiguous (each 8MB, 256B-aligned sizes) so the 32MB MLP
  // hidden buffer aliases them after attention is done.
  unsigned short* h1b = (unsigned short*)carve((size_t)BT * C * 2);
  unsigned short* qb  = (unsigned short*)carve((size_t)BT * C * 2);
  unsigned short* kb  = (unsigned short*)carve((size_t)BT * C * 2);
  unsigned short* vtb = (unsigned short*)carve((size_t)BT * C * 2);
  unsigned short* f1b = h1b;  // [BT, DFF] bf16, aliases h1b/qb/kb/vtb

  // weight transposes -> bf16 N-major
  transpose_cvt<<<dim3(D / 32, C / 32, H), 256, 0, stream>>>(Wq, WqT, C, D);
  transpose_cvt<<<dim3(D / 32, C / 32, H), 256, 0, stream>>>(Wk, WkT, C, D);
  transpose_cvt<<<dim3(D / 32, C / 32, H), 256, 0, stream>>>(Wv, WvT, C, D);
  transpose_cvt<<<dim3(C / 32, C / 32, 1), 256, 0, stream>>>(Wp, WpT, C, C);
  transpose_cvt<<<dim3(DFF / 32, C / 32, 1), 256, 0, stream>>>(W1, W1T, C, DFF);
  transpose_cvt<<<dim3(C / 32, DFF / 32, 1), 256, 0, stream>>>(W2, W2T, DFF, C);

  // LN1
  ln_kernel<<<BT, 256, 0, stream>>>(x, ln1_g, ln1_b, h1b);

  // QKV projections (N = H*D = 1024); grid = (M/128) * (N/64)
  constexpr int g1 = (BT / 128) * (1024 / 64);  // 512 blocks
  gemm_bf16<BT, 1024, 1024, EPI_QK><<<g1, 256, 0, stream>>>(h1b, WqT, nullptr, nullptr, qb);
  gemm_bf16<BT, 1024, 1024, EPI_QK><<<g1, 256, 0, stream>>>(h1b, WkT, nullptr, nullptr, kb);
  gemm_bf16<BT, 1024, 1024, EPI_VT><<<g1, 256, 0, stream>>>(h1b, WvT, nullptr, nullptr, vtb);

  // flash attention: B*H*(T/16) = 4096 waves -> 512 blocks
  attn_kernel<<<(B * H * (T / 16)) / 8, 256, 0, stream>>>(qb, kb, vtb, ob);

  // output projection + bias + residual -> x2 (fp32)
  gemm_bf16<BT, 1024, 1024, EPI_BIAS_RES_F32><<<g1, 256, 0, stream>>>(ob, WpT, bp, x, x2);

  // LN2
  ln_kernel<<<BT, 256, 0, stream>>>(x2, ln2_g, ln2_b, h2b);

  // MLP
  constexpr int g2 = (BT / 128) * (DFF / 64);  // 2048 blocks
  gemm_bf16<BT, DFF, 1024, EPI_RELU_BF16><<<g2, 256, 0, stream>>>(h2b, W1T, b1, nullptr, f1b);
  gemm_bf16<BT, 1024, DFF, EPI_BIAS_RES_F32><<<g1, 256, 0, stream>>>(f1b, W2T, b2, x2, out);
}
